// HybridKernelLayer_65481071407811
// MI455X (gfx1250) — matrix operand order repaired and checked by reading, not compiled, not run on hardware
//
#include <hip/hip_runtime.h>
#include <hip/hip_bf16.h>

// RBF kernel layer, fused:  out[n] = b + sum_m exp(-g*max(x2[n]+y2[m]-2*X[n].Y[m],0)) * W[m]
// Strategy:
//  1) cvt_kernel:   X,Y (f32) -> bf16 hi/lo split arrays (Markidis-style, ~f32 accuracy)
//  2) norms_kernel: x2[n], y2[m]
//  3) rbf_gemm:     v_wmma_f32_16x16x32_bf16 x3 (hi*hi + hi*lo + lo*hi) with fused exp/W epilogue
//  4) finish:       combine the 4 deterministic m-split partials + bias
// Workspace use (~49.4 MB): x2 64KB | y2 32KB | partial 256KB | Xhi 16MB | Xlo 16MB | Yhi 8MB | Ylo 8MB

#define N_ROWS 16384
#define M_COLS 8192
#define DDIM   512
#define GAMMA  0.002f
#define MSPLIT 4

typedef __attribute__((ext_vector_type(16))) __bf16 v16bf;
typedef __attribute__((ext_vector_type(8)))  __bf16 v8bf;
typedef __attribute__((ext_vector_type(8)))  float  v8f;

// ---------- bf16 helpers (bit-level, no dependence on __bf16 cast semantics) ----------
static __device__ __forceinline__ unsigned short f32_to_bf16_rne(float f) {
  unsigned int u = __float_as_uint(f);
  unsigned int lsb = (u >> 16) & 1u;
  u += 0x7FFFu + lsb;              // round-to-nearest-even (finite inputs)
  return (unsigned short)(u >> 16);
}
static __device__ __forceinline__ float bf16_bits_to_f32(unsigned short h) {
  return __uint_as_float(((unsigned int)h) << 16);
}

// ---------- fragment loaders (this code's contiguous k = 16h+s order for A and B, wave32) ----------
// A 16x32 bf16: lane L -> row L&15; half h=L>>4: K run {ks+16h .. ks+16h+15} (contiguous)
static __device__ __forceinline__ v16bf load_a_frag(const __bf16* p /* row + ks + 16*half */) {
  v8bf lo = *(const v8bf*)(p);
  v8bf hi = *(const v8bf*)(p + 8);
  v16bf r;
#pragma unroll
  for (int i = 0; i < 8; ++i) { r[i] = lo[i]; r[i + 8] = hi[i]; }
  return r;
}
// B 32x16 bf16: lane L -> col L&15; half h: K run {ks+16h .. ks+16h+15} (contiguous, 32B aligned)
static __device__ __forceinline__ v16bf load_b_frag(const __bf16* p /* row + ks + 16*half */) {
  return *(const v16bf*)(p);
}

// ---------- pass 1: split f32 -> bf16 hi/lo ----------
__global__ void cvt_kernel(const float* __restrict__ X, const float* __restrict__ Y,
                           unsigned short* __restrict__ Xh, unsigned short* __restrict__ Xl,
                           unsigned short* __restrict__ Yh, unsigned short* __restrict__ Yl) {
  size_t i = (size_t)blockIdx.x * blockDim.x + threadIdx.x;
  const size_t nx  = (size_t)N_ROWS * DDIM;
  const size_t tot = nx + (size_t)M_COLS * DDIM;
  if (i >= tot) return;
  float v; unsigned short *ph, *pl; size_t j;
  if (i < nx) { j = i;      v = X[j]; ph = Xh; pl = Xl; }
  else        { j = i - nx; v = Y[j]; ph = Yh; pl = Yl; }
  unsigned short h = f32_to_bf16_rne(v);
  ph[j] = h;
  pl[j] = f32_to_bf16_rne(v - bf16_bits_to_f32(h));
}

// ---------- pass 2: row norms ----------
__global__ void norms_kernel(const float* __restrict__ X, const float* __restrict__ Y,
                             float* __restrict__ x2, float* __restrict__ y2) {
  int r = blockIdx.x * blockDim.x + threadIdx.x;
  if (r >= N_ROWS + M_COLS) return;
  const float4* p = (r < N_ROWS) ? (const float4*)(X + (size_t)r * DDIM)
                                 : (const float4*)(Y + (size_t)(r - N_ROWS) * DDIM);
  float s = 0.f;
#pragma unroll 4
  for (int i = 0; i < DDIM / 4; ++i) {
    float4 v = p[i];
    s = fmaf(v.x, v.x, s); s = fmaf(v.y, v.y, s);
    s = fmaf(v.z, v.z, s); s = fmaf(v.w, v.w, s);
  }
  if (r < N_ROWS) x2[r] = s; else y2[r - N_ROWS] = s;
}

// ---------- pass 3: fused WMMA GEMM + exp + weighted row reduction ----------
// grid = (N_ROWS/64, MSPLIT), block = 128 (4 waves). Wave w owns rows blockIdx.x*64 + w*16 .. +15,
// scans its m-stripe in 32-col chunks (two 16x16 accumulators). All 4 waves share the m-stripe
// so B fragments hit the WGP cache. No atomics: m-splits land in distinct partial[] slices.
__global__ __launch_bounds__(128) void rbf_gemm_kernel(
    const __bf16* __restrict__ Xh, const __bf16* __restrict__ Xl,
    const __bf16* __restrict__ Yh, const __bf16* __restrict__ Yl,
    const float* __restrict__ x2, const float* __restrict__ y2,
    const float* __restrict__ W,  float* __restrict__ partial) {
  const int lane = threadIdx.x & 31;
  const int wave = threadIdx.x >> 5;
  const int half = lane >> 4;
  const int l16  = lane & 15;
  const int row_base = blockIdx.x * 64 + wave * 16;
  const int ms       = blockIdx.y;
  const int m_begin  = ms * (M_COLS / MSPLIT);
  const int m_end    = m_begin + (M_COLS / MSPLIT);

  const __bf16* xh_row = Xh + (size_t)(row_base + l16) * DDIM + 16 * half;
  const __bf16* xl_row = Xl + (size_t)(row_base + l16) * DDIM + 16 * half;

  float x2r[8];
#pragma unroll
  for (int v = 0; v < 8; ++v) x2r[v] = x2[row_base + v + half * 8];

  float wsum[8];
#pragma unroll
  for (int v = 0; v < 8; ++v) wsum[v] = 0.f;

  for (int m = m_begin; m < m_end; m += 32) {
    const __bf16* y0h = Yh + (size_t)(m + l16) * DDIM + 16 * half;
    const __bf16* y0l = Yl + (size_t)(m + l16) * DDIM + 16 * half;
    const __bf16* y1h = y0h + (size_t)16 * DDIM;
    const __bf16* y1l = y0l + (size_t)16 * DDIM;

    v8f acc0 = {}; v8f acc1 = {};
#pragma unroll 2
    for (int ks = 0; ks < DDIM; ks += 32) {
      v16bf ah  = load_a_frag(xh_row + ks);
      v16bf al  = load_a_frag(xl_row + ks);
      v16bf b0h = load_b_frag(y0h + ks);
      v16bf b0l = load_b_frag(y0l + ks);
      v16bf b1h = load_b_frag(y1h + ks);
      v16bf b1l = load_b_frag(y1l + ks);
      // xy ~= hi*hi + hi*lo + lo*hi  (f32 accumulate)
      acc0 = __builtin_amdgcn_wmma_f32_16x16x32_bf16(false, ah, false, b0h, (short)0, acc0, false, false);
      acc0 = __builtin_amdgcn_wmma_f32_16x16x32_bf16(false, ah, false, b0l, (short)0, acc0, false, false);
      acc0 = __builtin_amdgcn_wmma_f32_16x16x32_bf16(false, al, false, b0h, (short)0, acc0, false, false);
      acc1 = __builtin_amdgcn_wmma_f32_16x16x32_bf16(false, ah, false, b1h, (short)0, acc1, false, false);
      acc1 = __builtin_amdgcn_wmma_f32_16x16x32_bf16(false, ah, false, b1l, (short)0, acc1, false, false);
      acc1 = __builtin_amdgcn_wmma_f32_16x16x32_bf16(false, al, false, b1h, (short)0, acc1, false, false);
    }

    // fused epilogue: C layout -> lane holds col N = l16, VGPR v holds row M = v + 8*half
    const int c0 = m + l16, c1 = m + 16 + l16;
    const float w0 = W[c0], w1 = W[c1];
    const float yy0 = y2[c0], yy1 = y2[c1];
#pragma unroll
    for (int v = 0; v < 8; ++v) {
      float s0 = fmaxf(x2r[v] + yy0 - 2.0f * acc0[v], 0.0f);
      float s1 = fmaxf(x2r[v] + yy1 - 2.0f * acc1[v], 0.0f);
      wsum[v] = fmaf(__expf(-GAMMA * s0), w0, wsum[v]);
      wsum[v] = fmaf(__expf(-GAMMA * s1), w1, wsum[v]);
    }
  }

  // reduce over the 16 lanes sharing each row (xor masks < 16 stay within the half)
#pragma unroll
  for (int v = 0; v < 8; ++v) {
    float s = wsum[v];
    s += __shfl_xor(s, 1, 32);
    s += __shfl_xor(s, 2, 32);
    s += __shfl_xor(s, 4, 32);
    s += __shfl_xor(s, 8, 32);
    wsum[v] = s;
  }
  if (l16 == 0) {
#pragma unroll
    for (int v = 0; v < 8; ++v)
      partial[(size_t)ms * N_ROWS + row_base + v + half * 8] = wsum[v];
  }
}

// ---------- pass 4: combine m-split partials + bias ----------
__global__ void finish_kernel(const float* __restrict__ partial, const float* __restrict__ b,
                              float* __restrict__ out) {
  int i = blockIdx.x * blockDim.x + threadIdx.x;
  if (i >= N_ROWS) return;
  float s = b[0];
#pragma unroll
  for (int sp = 0; sp < MSPLIT; ++sp) s += partial[(size_t)sp * N_ROWS + i];
  out[i] = s;
}

extern "C" void kernel_launch(void* const* d_in, const int* in_sizes, int n_in,
                              void* d_out, int out_size, void* d_ws, size_t ws_size,
                              hipStream_t stream) {
  const float* X = (const float*)d_in[0];   // (16384, 512)
  const float* Y = (const float*)d_in[1];   // (8192, 512)
  const float* W = (const float*)d_in[2];   // (1, 8192)
  const float* b = (const float*)d_in[3];   // (1,)
  float* out = (float*)d_out;               // (16384,)

  char* ws = (char*)d_ws;
  float* x2      = (float*)(ws);                                  //  64 KB
  float* y2      = (float*)(ws + 65536);                          //  32 KB
  float* partial = (float*)(ws + 98304);                          // 256 KB
  const size_t XH = (size_t)1 << 20;                              // 1 MB aligned region start
  unsigned short* Xh = (unsigned short*)(ws + XH);                                   // 16 MB
  unsigned short* Xl = (unsigned short*)(ws + XH + (size_t)N_ROWS * DDIM * 2);       // 16 MB
  unsigned short* Yh = (unsigned short*)(ws + XH + (size_t)N_ROWS * DDIM * 4);       //  8 MB
  unsigned short* Yl = (unsigned short*)(ws + XH + (size_t)N_ROWS * DDIM * 4
                                            + (size_t)M_COLS * DDIM * 2);            //  8 MB

  const size_t tot = (size_t)(N_ROWS + M_COLS) * DDIM;
  cvt_kernel<<<(unsigned)((tot + 255) / 256), 256, 0, stream>>>(X, Y, Xh, Xl, Yh, Yl);
  norms_kernel<<<(N_ROWS + M_COLS + 255) / 256, 256, 0, stream>>>(X, Y, x2, y2);
  rbf_gemm_kernel<<<dim3(N_ROWS / 64, MSPLIT), 128, 0, stream>>>(
      (const __bf16*)Xh, (const __bf16*)Xl, (const __bf16*)Yh, (const __bf16*)Yl,
      x2, y2, W, partial);
  finish_kernel<<<N_ROWS / 256, 256, 0, stream>>>(partial, b, out);
}
